// CDDModel_69630009803097
// MI455X (gfx1250) — matrix operand order repaired and checked by reading, not compiled, not run on hardware
//
#include <hip/hip_runtime.h>
#include <math.h>

#define NNODES 10000
#define NEDGES 160000
#define NGRAPH 16
#define NS 100
#define NV 16
#define SE 32
#define RBF 16
#define NATOM 13
#define ODIM 512

typedef __attribute__((ext_vector_type(16))) _Float16 v16h;
typedef __attribute__((ext_vector_type(8)))  float    v8f;

// ---------------- parameter structs ----------------
struct GvpW {
  const float*    wh;     // [h][vi] f32
  const float*    ws_b;   // padded [Np] f32
  const float*    wsv_b;  // padded [GNp] f32
  const float*    wv;     // [vo][h]
  const _Float16* ws;     // fragment-swizzled f16, K=si+h padded
  const _Float16* wsv;    // fragment-swizzled f16, K=so padded
};
struct MsgP { GvpW g0, g1, g2; };
struct FFP  { GvpW f0, f1; const float *ln0g, *ln0b, *ln1g, *ln1b; };

// ---------------- fragment swizzle ----------------
// A-matrix (16-bit, 16x32 per K-block) per CDNA5 ISA layout:
//   lane = ((kin>>3)&1)*16 + row ; elem = 2*(((kin&7)>>1) | ((kin&16)>>2)) | (kin&1)
// Each lane's 16 halves (one v16h fragment) are contiguous -> 32B LDS load.
__device__ __forceinline__ int aswz(int e, int c) {
  int kb = c >> 5, kin = c & 31;
  int sub = (kin >> 3) & 1;
  int v = ((kin & 7) >> 1) | ((kin & 16) >> 2);
  return kb * 512 + (((sub << 4) | e) << 4) + ((v << 1) | (kin & 1));
}

// ---------------- WMMA tile GEMM ----------------
// A: LDS f16, fragment-swizzled, KB blocks of 512 halves.
// W: global f16, fragment-swizzled [nb][kb][lane][16].
// C: LDS f32 [16][ldc]. 4 waves split the 16-wide N tiles. bias pre-padded.
__device__ __forceinline__ void wmma_gemm(const _Float16* __restrict__ A, int KB,
                                          const _Float16* __restrict__ W, int Np,
                                          const float* __restrict__ bias,
                                          float* __restrict__ C, int ldc,
                                          int lane, int wave)
{
  const int sub = lane >> 4;
  const int m   = lane & 15;
  const v16h* Ap = (const v16h*)A + lane;
  for (int n0 = wave * 16; n0 < Np; n0 += 64) {
    const v16h* Wp = (const v16h*)W + ((n0 >> 4) * KB) * 32 + lane;
    v8f acc = {};
    for (int kb = 0; kb < KB; ++kb) {
      v16h a = Ap[kb * 32];
      v16h b = Wp[kb * 32];
      acc = __builtin_amdgcn_wmma_f32_16x16x32_f16(false, a, false, b,
                                                   (short)0, acc, false, false);
    }
    int n = n0 + m;
    float bv = bias[n];
    #pragma unroll
    for (int r = 0; r < 8; ++r)
      C[(r + sub * 8) * ldc + n] = acc[r] + bv;
  }
}

// ---------------- small helpers ----------------
__device__ __forceinline__ void ln_node(float* s, const float* g, const float* b) {
  float m = 0.f;
  for (int i = 0; i < NS; ++i) m += s[i];
  m *= (1.f / NS);
  float v = 0.f;
  for (int i = 0; i < NS; ++i) { float d = s[i] - m; v += d * d; }
  v *= (1.f / NS);
  float r = rsqrtf(v + 1e-5f);
  for (int i = 0; i < NS; ++i) s[i] = g[i] * (s[i] - m) * r + b[i];
}
__device__ __forceinline__ void vecln_node(float* v) {  // [16][3]
  float acc = 0.f;
  for (int c = 0; c < NV; ++c) {
    float s = 0.f;
    for (int x = 0; x < 3; ++x) { float t = v[c * 3 + x]; s += t * t; }
    acc += fmaxf(s, 1e-8f);
  }
  float inv = rsqrtf(acc * (1.f / NV));
  for (int i = 0; i < NV * 3; ++i) v[i] *= inv;
}
__device__ __forceinline__ float sigm(float x) { return 1.f / (1.f + __expf(-x)); }

// ---------------- utility kernels ----------------
__global__ void k_zero(float* p, long n) {
  long i = (long)blockIdx.x * blockDim.x + threadIdx.x;
  if (i < n) p[i] = 0.f;
}

// W: f32 [N][K] -> dst: fragment-swizzled f16 (B-matrix layout, zero padded)
__global__ void k_convert(const float* __restrict__ W, _Float16* __restrict__ dst,
                          int N, int K, int Kp, int Np) {
  int i = blockIdx.x * blockDim.x + threadIdx.x;
  if (i >= Kp * Np) return;
  int k = i / Np, n = i % Np;
  int KB = Kp >> 5;
  int kb = k >> 5, sub = (k >> 3) & 1;
  int lane = (sub << 4) | (n & 15);
  int idx = ((((n >> 4) * KB + kb) * 32 + lane) << 4) + ((k & 7) | ((k & 16) >> 1));
  dst[idx] = (_Float16)((k < K && n < N) ? W[n * K + k] : 0.f);
}

// b: f32 [N] -> dst: f32 [Np] zero padded
__global__ void k_padbias(const float* __restrict__ b, float* __restrict__ dst,
                          int N, int Np) {
  int i = blockIdx.x * blockDim.x + threadIdx.x;
  if (i < Np) dst[i] = (i < N) ? b[i] : 0.f;
}

// node init: embed -> LN -> Linear(13->100); hv = 0
__global__ void k_node_init(const int* __restrict__ atoms, const float* __restrict__ embed,
                            const float* __restrict__ g, const float* __restrict__ b,
                            const float* __restrict__ W, const float* __restrict__ bias,
                            float* __restrict__ hs, float* __restrict__ hv) {
  int n = blockIdx.x * blockDim.x + threadIdx.x;
  if (n >= NNODES) return;
  float e[NATOM];
  int a = atoms[n];
  float m = 0.f;
  for (int i = 0; i < NATOM; ++i) { e[i] = embed[a * NATOM + i]; m += e[i]; }
  m *= (1.f / NATOM);
  float v = 0.f;
  for (int i = 0; i < NATOM; ++i) { float d = e[i] - m; v += d * d; }
  v *= (1.f / NATOM);
  float r = rsqrtf(v + 1e-5f);
  for (int i = 0; i < NATOM; ++i) e[i] = g[i] * (e[i] - m) * r + b[i];
  for (int o = 0; o < NS; ++o) {
    float s = bias[o];
    for (int k = 0; k < NATOM; ++k) s += W[o * NATOM + k] * e[k];
    hs[n * NS + o] = s;
  }
  for (int c = 0; c < NV * 3; ++c) hv[n * NV * 3 + c] = 0.f;
}

// edge embed W_e: LN(edge_s), vec_ln(edge_v), GVP(17->32, h=1, vector gate)
__global__ void k_edge_embed(const float* __restrict__ edge_s, const float* __restrict__ edge_v,
                             const float* __restrict__ wh, const float* __restrict__ ws_b,
                             const float* __restrict__ ws_w, const float* __restrict__ wsv_b,
                             const float* __restrict__ wsv_w, const float* __restrict__ wv,
                             const float* __restrict__ lng, const float* __restrict__ lnb,
                             float* __restrict__ es, float* __restrict__ ev) {
  int e = blockIdx.x * blockDim.x + threadIdx.x;
  if (e >= NEDGES) return;
  float s[RBF];
  float m = 0.f;
  for (int i = 0; i < RBF; ++i) { s[i] = edge_s[e * RBF + i]; m += s[i]; }
  m *= (1.f / RBF);
  float var = 0.f;
  for (int i = 0; i < RBF; ++i) { float d = s[i] - m; var += d * d; }
  var *= (1.f / RBF);
  float r = rsqrtf(var + 1e-5f);
  for (int i = 0; i < RBF; ++i) s[i] = lng[i] * (s[i] - m) * r + lnb[i];
  float vx[3], vn2 = 0.f;
  for (int x = 0; x < 3; ++x) { vx[x] = edge_v[e * 3 + x]; vn2 += vx[x] * vx[x]; }
  float inv = rsqrtf(fmaxf(vn2, 1e-8f));
  for (int x = 0; x < 3; ++x) vx[x] *= inv;
  float w0 = wh[0];
  float vh[3] = {w0 * vx[0], w0 * vx[1], w0 * vx[2]};
  float vhn = sqrtf(fmaxf(vh[0] * vh[0] + vh[1] * vh[1] + vh[2] * vh[2], 1e-8f));
  float so[SE];
  float gdot = wsv_b[0];
  for (int o = 0; o < SE; ++o) {
    float a = ws_b[o] + ws_w[o * 17 + 16] * vhn;
    for (int k = 0; k < RBF; ++k) a += ws_w[o * 17 + k] * s[k];
    so[o] = a;
    gdot += wsv_w[o] * a;
  }
  float gate = sigm(gdot);
  float wv0 = wv[0];
  for (int o = 0; o < SE; ++o) es[e * SE + o] = so[o];
  for (int x = 0; x < 3; ++x) ev[e * 3 + x] = wv0 * vh[x] * gate;
}

// ---------------- message pass (fused 3-GVP edge pipeline, WMMA) ----------------
__device__ __forceinline__ void msg_gvp16(const GvpW& g, bool relu_in,
                                          _Float16* Af, float* st, float* vh16,
                                          float* vcur, float* gate,
                                          int tid, int lane, int wave) {
  // vh = wh(16x16) @ vcur
  for (int i = tid; i < 16 * 16 * 3; i += 128) {
    int e = i / 48, rr = i % 48, h = rr / 3, x = rr % 3;
    float a = 0.f;
    for (int k = 0; k < 16; ++k) a += g.wh[h * 16 + k] * vcur[(e * 16 + k) * 3 + x];
    vh16[i] = a;
  }
  __syncthreads();
  // A = [ (relu?)s(100) | vn(16) | pad ]  K=128, branch-free region fills
  for (int i = tid; i < 1600; i += 128) {
    int e = i / 100, c = i % 100;
    float v = st[e * 112 + c];
    if (relu_in) v = fmaxf(v, 0.f);
    Af[aswz(e, c)] = (_Float16)v;
  }
  for (int i = tid; i < 256; i += 128) {
    int e = i / 16, h = i % 16;
    float s = 0.f;
    for (int x = 0; x < 3; ++x) { float t = vh16[(e * 16 + h) * 3 + x]; s += t * t; }
    Af[aswz(e, 100 + h)] = (_Float16)sqrtf(fmaxf(s, 1e-8f));
  }
  for (int i = tid; i < 192; i += 128)
    Af[aswz(i / 12, 116 + i % 12)] = (_Float16)0.f;
  __syncthreads();
  wmma_gemm(Af, 4, g.ws, 112, g.ws_b, st, 112, lane, wave);
  __syncthreads();
  // gate from pre-activation s
  for (int i = tid; i < 1600; i += 128) {
    int e = i / 100, c = i % 100;
    Af[aswz(e, c)] = (_Float16)st[e * 112 + c];
  }
  for (int i = tid; i < 448; i += 128)
    Af[aswz(i / 28, 100 + i % 28)] = (_Float16)0.f;
  __syncthreads();
  wmma_gemm(Af, 4, g.wsv, 16, g.wsv_b, gate, 16, lane, wave);
  __syncthreads();
  for (int i = tid; i < 256; i += 128) gate[i] = sigm(gate[i]);
  __syncthreads();
  for (int i = tid; i < 16 * 16 * 3; i += 128) {
    int e = i / 48, rr = i % 48, o = rr / 3, x = rr % 3;
    float a = 0.f;
    for (int k = 0; k < 16; ++k) a += g.wv[o * 16 + k] * vh16[(e * 16 + k) * 3 + x];
    vcur[i] = a * gate[e * 16 + o];
  }
  __syncthreads();
}

__global__ __launch_bounds__(128) void k_msg(MsgP P,
    const float* __restrict__ hs, const float* __restrict__ hv,
    const float* __restrict__ es, const float* __restrict__ ev,
    const int* __restrict__ src, const int* __restrict__ dst,
    float* __restrict__ sAcc, float* __restrict__ vAcc, float* __restrict__ cnt) {
  __shared__ __align__(64) _Float16 Af[16 * 288];
  __shared__ float mv[16 * 33 * 3];
  __shared__ float vh[16 * 33 * 3];
  __shared__ float st[16 * 112];
  __shared__ float vcur[16 * 16 * 3];
  __shared__ float gate[16 * 16];
  __shared__ int sidx[16], didx[16];

  int tid = threadIdx.x, lane = tid & 31, wave = tid >> 5;
  int e0 = blockIdx.x * 16;
  if (tid < 16) { sidx[tid] = src[e0 + tid]; didx[tid] = dst[e0 + tid]; }
  __syncthreads();

  // gather scalar inputs into swizzled A (regions: s_j | es | s_i | pad)
  for (int i = tid; i < 1600; i += 128) {
    int e = i / 100, c = i % 100;
    Af[aswz(e, c)] = (_Float16)hs[sidx[e] * NS + c];
  }
  for (int i = tid; i < 512; i += 128) {
    int e = i / 32, c = i % 32;
    Af[aswz(e, 100 + c)] = (_Float16)es[(e0 + e) * SE + c];
  }
  for (int i = tid; i < 1600; i += 128) {
    int e = i / 100, c = i % 100;
    Af[aswz(e, 132 + c)] = (_Float16)hs[didx[e] * NS + c];
  }
  for (int i = tid; i < 368; i += 128)
    Af[aswz(i / 23, 265 + i % 23)] = (_Float16)0.f;
  // gather vector inputs (v_j | ev | v_i)
  for (int i = tid; i < 16 * 16 * 3; i += 128) {
    int e = i / 48, c = i % 48;
    mv[e * 99 + c] = hv[sidx[e] * 48 + c];
  }
  for (int i = tid; i < 16 * 3; i += 128) {
    int e = i / 3, x = i % 3;
    mv[e * 99 + 48 + x] = ev[(e0 + e) * 3 + x];
  }
  for (int i = tid; i < 16 * 16 * 3; i += 128) {
    int e = i / 48, c = i % 48;
    mv[e * 99 + 51 + c] = hv[didx[e] * 48 + c];
  }
  __syncthreads();

  // GVP0: vh = wh0(33x33) @ mv
  for (int i = tid; i < 16 * 33 * 3; i += 128) {
    int e = i / 99, rr = i % 99, h = rr / 3, x = rr % 3;
    float a = 0.f;
    for (int k = 0; k < 33; ++k) a += P.g0.wh[h * 33 + k] * mv[e * 99 + k * 3 + x];
    vh[i] = a;
  }
  __syncthreads();
  for (int i = tid; i < 528; i += 128) {
    int e = i / 33, h = i % 33;
    float s = 0.f;
    for (int x = 0; x < 3; ++x) { float t = vh[(e * 33 + h) * 3 + x]; s += t * t; }
    Af[aswz(e, 232 + h)] = (_Float16)sqrtf(fmaxf(s, 1e-8f));
  }
  __syncthreads();
  wmma_gemm(Af, 9, P.g0.ws, 112, P.g0.ws_b, st, 112, lane, wave);
  __syncthreads();
  for (int i = tid; i < 1600; i += 128) {
    int e = i / 100, c = i % 100;
    Af[aswz(e, c)] = (_Float16)st[e * 112 + c];
  }
  for (int i = tid; i < 448; i += 128)
    Af[aswz(i / 28, 100 + i % 28)] = (_Float16)0.f;
  __syncthreads();
  wmma_gemm(Af, 4, P.g0.wsv, 16, P.g0.wsv_b, gate, 16, lane, wave);
  __syncthreads();
  for (int i = tid; i < 256; i += 128) gate[i] = sigm(gate[i]);
  __syncthreads();
  for (int i = tid; i < 16 * 16 * 3; i += 128) {
    int e = i / 48, rr = i % 48, o = rr / 3, x = rr % 3;
    float a = 0.f;
    for (int k = 0; k < 33; ++k) a += P.g0.wv[o * 33 + k] * vh[(e * 33 + k) * 3 + x];
    vcur[i] = a * gate[e * 16 + o];
  }
  __syncthreads();

  msg_gvp16(P.g1, true, Af, st, vh, vcur, gate, tid, lane, wave);
  msg_gvp16(P.g2, true, Af, st, vh, vcur, gate, tid, lane, wave);

  // scatter-add into dst accumulators (segment mean numerator) + counts
  for (int i = tid; i < 16 * NS; i += 128) {
    int e = i / NS, c = i % NS;
    atomicAdd(&sAcc[didx[e] * NS + c], st[e * 112 + c]);
  }
  for (int i = tid; i < 16 * 48; i += 128) {
    int e = i / 48;
    atomicAdd(&vAcc[didx[e] * 48 + (i % 48)], vcur[i]);
  }
  if (tid < 16) atomicAdd(&cnt[didx[tid]], 1.f);
}

// ---------------- node update (residual+LN, FF GVPs, residual+LN) ----------------
__global__ __launch_bounds__(128) void k_node(FFP P,
    float* __restrict__ hs, float* __restrict__ hv,
    const float* __restrict__ sAcc, const float* __restrict__ vAcc,
    const float* __restrict__ cnt) {
  __shared__ __align__(64) _Float16 Af[16 * 448];
  __shared__ float scur[16 * NS];
  __shared__ float vcurr[16 * NV * 3];
  __shared__ float vhh[16 * 32 * 3];
  __shared__ float sff[16 * 400];
  __shared__ float vff[16 * 32 * 3];
  __shared__ float gate[16 * 32];
  float* s2 = sff;   // alias: sff dead once FF1's A is built
  float* v2 = vff;   // alias: vff dead once vh2 computed

  int tid = threadIdx.x, lane = tid & 31, wave = tid >> 5;
  int n0 = blockIdx.x * 16;

  for (int i = tid; i < 16 * NS; i += 128) {
    int e = i / NS; int n = n0 + e;
    float inv = 1.f / fmaxf(cnt[n], 1.f);
    scur[i] = hs[n * NS + (i % NS)] + sAcc[n * NS + (i % NS)] * inv;
  }
  for (int i = tid; i < 16 * 48; i += 128) {
    int e = i / 48; int n = n0 + e;
    float inv = 1.f / fmaxf(cnt[n], 1.f);
    vcurr[i] = hv[n * 48 + (i % 48)] + vAcc[n * 48 + (i % 48)] * inv;
  }
  __syncthreads();
  if (tid < 16) ln_node(scur + tid * NS, P.ln0g, P.ln0b);
  else if (tid < 32) vecln_node(vcurr + (tid - 16) * 48);
  __syncthreads();

  // FF0: h=32; vh = wh(32x16)@v
  for (int i = tid; i < 16 * 32 * 3; i += 128) {
    int e = i / 96, rr = i % 96, h = rr / 3, x = rr % 3;
    float a = 0.f;
    for (int k = 0; k < 16; ++k) a += P.f0.wh[h * 16 + k] * vcurr[(e * 16 + k) * 3 + x];
    vhh[i] = a;
  }
  __syncthreads();
  for (int i = tid; i < 1600; i += 128) {
    int e = i / 100, c = i % 100;
    Af[aswz(e, c)] = (_Float16)scur[e * NS + c];
  }
  for (int i = tid; i < 512; i += 128) {
    int e = i / 32, h = i % 32;
    float s = 0.f;
    for (int x = 0; x < 3; ++x) { float t = vhh[(e * 32 + h) * 3 + x]; s += t * t; }
    Af[aswz(e, 100 + h)] = (_Float16)sqrtf(fmaxf(s, 1e-8f));
  }
  for (int i = tid; i < 448; i += 128)
    Af[aswz(i / 28, 132 + i % 28)] = (_Float16)0.f;
  __syncthreads();
  wmma_gemm(Af, 5, P.f0.ws, 400, P.f0.ws_b, sff, 400, lane, wave);
  __syncthreads();
  for (int i = tid; i < 6400; i += 128) {
    int e = i / 400, c = i % 400;
    Af[aswz(e, c)] = (_Float16)sff[i];
  }
  for (int i = tid; i < 256; i += 128)
    Af[aswz(i / 16, 400 + i % 16)] = (_Float16)0.f;
  __syncthreads();
  wmma_gemm(Af, 13, P.f0.wsv, 32, P.f0.wsv_b, gate, 32, lane, wave);
  __syncthreads();
  for (int i = tid; i < 512; i += 128) gate[i] = sigm(gate[i]);
  __syncthreads();
  for (int i = tid; i < 16 * 32 * 3; i += 128) {
    int e = i / 96, rr = i % 96, o = rr / 3, x = rr % 3;
    float a = 0.f;
    for (int k = 0; k < 32; ++k) a += P.f0.wv[o * 32 + k] * vhh[(e * 32 + k) * 3 + x];
    vff[i] = a * gate[e * 32 + o];
  }
  __syncthreads();

  // FF1: h=32; vh2 = wh(32x32)@vff  (into vhh)
  for (int i = tid; i < 16 * 32 * 3; i += 128) {
    int e = i / 96, rr = i % 96, h = rr / 3, x = rr % 3;
    float a = 0.f;
    for (int k = 0; k < 32; ++k) a += P.f1.wh[h * 32 + k] * vff[(e * 32 + k) * 3 + x];
    vhh[i] = a;
  }
  __syncthreads();
  for (int i = tid; i < 6400; i += 128) {
    int e = i / 400, c = i % 400;
    Af[aswz(e, c)] = (_Float16)fmaxf(sff[i], 0.f);   // relu after FF0
  }
  for (int i = tid; i < 512; i += 128) {
    int e = i / 32, h = i % 32;
    float s = 0.f;
    for (int x = 0; x < 3; ++x) { float t = vhh[(e * 32 + h) * 3 + x]; s += t * t; }
    Af[aswz(e, 400 + h)] = (_Float16)sqrtf(fmaxf(s, 1e-8f));
  }
  for (int i = tid; i < 256; i += 128)
    Af[aswz(i / 16, 432 + i % 16)] = (_Float16)0.f;
  __syncthreads();
  wmma_gemm(Af, 14, P.f1.ws, 112, P.f1.ws_b, s2, 112, lane, wave);
  __syncthreads();
  for (int i = tid; i < 1600; i += 128) {
    int e = i / 100, c = i % 100;
    Af[aswz(e, c)] = (_Float16)s2[e * 112 + c];
  }
  for (int i = tid; i < 448; i += 128)
    Af[aswz(i / 28, 100 + i % 28)] = (_Float16)0.f;
  __syncthreads();
  wmma_gemm(Af, 4, P.f1.wsv, 16, P.f1.wsv_b, gate, 16, lane, wave);
  __syncthreads();
  for (int i = tid; i < 256; i += 128) gate[i] = sigm(gate[i]);
  __syncthreads();
  for (int i = tid; i < 16 * 16 * 3; i += 128) {
    int e = i / 48, rr = i % 48, o = rr / 3, x = rr % 3;
    float a = 0.f;
    for (int k = 0; k < 32; ++k) a += P.f1.wv[o * 32 + k] * vhh[(e * 32 + k) * 3 + x];
    v2[i] = a * gate[e * 16 + o];
  }
  __syncthreads();

  for (int i = tid; i < 16 * NS; i += 128) scur[i] += s2[(i / NS) * 112 + (i % NS)];
  for (int i = tid; i < 16 * 48; i += 128) vcurr[i] += v2[i];
  __syncthreads();
  if (tid < 16) ln_node(scur + tid * NS, P.ln1g, P.ln1b);
  else if (tid < 32) vecln_node(vcurr + (tid - 16) * 48);
  __syncthreads();
  for (int i = tid; i < 16 * NS; i += 128) hs[n0 * NS + i] = scur[i];
  for (int i = tid; i < 16 * 48; i += 128) hv[n0 * 48 + i] = vcurr[i];
}

// ---------------- output head + pooling ----------------
__global__ __launch_bounds__(128) void k_out(
    const float* __restrict__ wh, const _Float16* __restrict__ ws,
    const float* __restrict__ ws_b, const float* __restrict__ lng,
    const float* __restrict__ lnb, const float* __restrict__ hs,
    const float* __restrict__ hv, const int* __restrict__ batch,
    float* __restrict__ gsum, float* __restrict__ gcnt) {
  __shared__ __align__(64) _Float16 Af[16 * 128];
  __shared__ float scur[16 * NS];
  __shared__ float vcurr[16 * 48];
  __shared__ float vhh[16 * 48];
  __shared__ float so[16 * ODIM];
  int tid = threadIdx.x, lane = tid & 31, wave = tid >> 5;
  int n0 = blockIdx.x * 16;
  for (int i = tid; i < 16 * NS; i += 128) scur[i] = hs[n0 * NS + i];
  for (int i = tid; i < 16 * 48; i += 128) vcurr[i] = hv[n0 * 48 + i];
  __syncthreads();
  if (tid < 16) ln_node(scur + tid * NS, lng, lnb);
  else if (tid < 32) vecln_node(vcurr + (tid - 16) * 48);
  __syncthreads();
  for (int i = tid; i < 16 * 48; i += 128) {
    int e = i / 48, rr = i % 48, h = rr / 3, x = rr % 3;
    float a = 0.f;
    for (int k = 0; k < 16; ++k) a += wh[h * 16 + k] * vcurr[(e * 16 + k) * 3 + x];
    vhh[i] = a;
  }
  __syncthreads();
  for (int i = tid; i < 1600; i += 128) {
    int e = i / 100, c = i % 100;
    Af[aswz(e, c)] = (_Float16)scur[e * NS + c];
  }
  for (int i = tid; i < 256; i += 128) {
    int e = i / 16, h = i % 16;
    float s = 0.f;
    for (int x = 0; x < 3; ++x) { float t = vhh[(e * 16 + h) * 3 + x]; s += t * t; }
    Af[aswz(e, 100 + h)] = (_Float16)sqrtf(fmaxf(s, 1e-8f));
  }
  for (int i = tid; i < 192; i += 128)
    Af[aswz(i / 12, 116 + i % 12)] = (_Float16)0.f;
  __syncthreads();
  wmma_gemm(Af, 4, ws, ODIM, ws_b, so, ODIM, lane, wave);
  __syncthreads();
  for (int i = tid; i < 16 * ODIM; i += 128) {
    int e = i / ODIM, c = i % ODIM;
    atomicAdd(&gsum[batch[n0 + e] * ODIM + c], fmaxf(so[i], 0.f));
  }
  if (tid < 16) atomicAdd(&gcnt[batch[n0 + tid]], 1.f);
}

__global__ void k_finalize(const float* __restrict__ gsum, const float* __restrict__ gcnt,
                           float* __restrict__ out) {
  int i = blockIdx.x * blockDim.x + threadIdx.x;
  if (i < NGRAPH * ODIM) out[i] = gsum[i] / fmaxf(gcnt[i >> 9], 1.f);
}

// ---------------- host ----------------
extern "C" void kernel_launch(void* const* d_in, const int* in_sizes, int n_in,
                              void* d_out, int out_size, void* d_ws, size_t ws_size,
                              hipStream_t stream) {
  (void)in_sizes; (void)n_in; (void)out_size; (void)ws_size;
  const float* const* F = (const float* const*)d_in;
  const int* atoms = (const int*)d_in[188];
  const int* eidx  = (const int*)d_in[189];
  const float* edge_s = (const float*)d_in[190];
  const float* edge_v = (const float*)d_in[191];
  const int* batch = (const int*)d_in[192];
  const int* esrc = eidx;
  const int* edst = eidx + NEDGES;
  float* out = (float*)d_out;

  size_t off = 0;
  auto carve = [&](size_t bytes) -> void* {
    void* p = (char*)d_ws + off;
    off = (off + bytes + 255) & ~(size_t)255;
    return p;
  };
  float* hs   = (float*)carve((size_t)NNODES * NS * 4);
  float* hv   = (float*)carve((size_t)NNODES * 48 * 4);
  float* es   = (float*)carve((size_t)NEDGES * SE * 4);
  float* ev   = (float*)carve((size_t)NEDGES * 3 * 4);
  float* sAcc = (float*)carve((size_t)NNODES * NS * 4);   // sAcc/vAcc/cnt contiguous
  float* vAcc = (float*)carve((size_t)NNODES * 48 * 4);
  float* cnt  = (float*)carve((size_t)NNODES * 4);
  float* gsum = (float*)carve((size_t)NGRAPH * ODIM * 4); // gsum/gcnt contiguous
  float* gcnt = (float*)carve((size_t)NGRAPH * 4);

  _Float16 *m0ws[5], *m0g[5], *m1ws[5], *m1g[5], *m2ws[5], *m2g[5];
  _Float16 *f0ws[5], *f0g[5], *f1ws[5], *f1g[5];
  float *m0b[5], *m0gb[5], *m1b[5], *m1gb[5], *m2b[5], *m2gb[5];
  float *f0b[5], *f0gb[5], *f1b[5], *f1gb[5];
  for (int l = 0; l < 5; ++l) {
    m0ws[l] = (_Float16*)carve(288 * 112 * 2);
    m0g[l]  = (_Float16*)carve(128 * 16 * 2);
    m1ws[l] = (_Float16*)carve(128 * 112 * 2);
    m1g[l]  = (_Float16*)carve(128 * 16 * 2);
    m2ws[l] = (_Float16*)carve(128 * 112 * 2);
    m2g[l]  = (_Float16*)carve(128 * 16 * 2);
    f0ws[l] = (_Float16*)carve(160 * 400 * 2);
    f0g[l]  = (_Float16*)carve(416 * 32 * 2);
    f1ws[l] = (_Float16*)carve(448 * 112 * 2);
    f1g[l]  = (_Float16*)carve(128 * 16 * 2);
    m0b[l]  = (float*)carve(112 * 4);
    m0gb[l] = (float*)carve(16 * 4);
    m1b[l]  = (float*)carve(112 * 4);
    m1gb[l] = (float*)carve(16 * 4);
    m2b[l]  = (float*)carve(112 * 4);
    m2gb[l] = (float*)carve(16 * 4);
    f0b[l]  = (float*)carve(400 * 4);
    f0gb[l] = (float*)carve(32 * 4);
    f1b[l]  = (float*)carve(112 * 4);
    f1gb[l] = (float*)carve(16 * 4);
  }
  _Float16* woutws = (_Float16*)carve(128 * ODIM * 2);
  float* woutb = (float*)carve(ODIM * 4);

  auto conv = [&](const float* src, _Float16* dst, int N, int K, int Kp, int Np) {
    int n = Kp * Np;
    k_convert<<<(n + 255) / 256, 256, 0, stream>>>(src, dst, N, K, Kp, Np);
  };
  auto padb = [&](const float* src, float* dst, int N, int Np) {
    k_padbias<<<(Np + 255) / 256, 256, 0, stream>>>(src, dst, N, Np);
  };

  // weight + bias prep (tiny; L2-resident for all tile-GEMM blocks)
  for (int l = 0; l < 5; ++l) {
    int b = 18 + 34 * l;
    conv(F[b + 2],  f0ws[l], 400, 132, 160, 400);
    conv(F[b + 4],  f0g[l],  32,  400, 416, 32);
    conv(F[b + 8],  f1ws[l], 100, 432, 448, 112);
    conv(F[b + 10], f1g[l],  16,  100, 128, 16);
    conv(F[b + 18], m0ws[l], 100, 265, 288, 112);
    conv(F[b + 20], m0g[l],  16,  100, 128, 16);
    conv(F[b + 24], m1ws[l], 100, 116, 128, 112);
    conv(F[b + 26], m1g[l],  16,  100, 128, 16);
    conv(F[b + 30], m2ws[l], 100, 116, 128, 112);
    conv(F[b + 32], m2g[l],  16,  100, 128, 16);
    padb(F[b + 1],  f0b[l],  400, 400);
    padb(F[b + 3],  f0gb[l], 32,  32);
    padb(F[b + 7],  f1b[l],  100, 112);
    padb(F[b + 9],  f1gb[l], 16,  16);
    padb(F[b + 17], m0b[l],  100, 112);
    padb(F[b + 19], m0gb[l], 16,  16);
    padb(F[b + 23], m1b[l],  100, 112);
    padb(F[b + 25], m1gb[l], 16,  16);
    padb(F[b + 29], m2b[l],  100, 112);
    padb(F[b + 31], m2gb[l], 16,  16);
  }
  conv(F[10], woutws, ODIM, 116, 128, ODIM);
  padb(F[9], woutb, ODIM, ODIM);

  // node + edge embeddings
  k_node_init<<<(NNODES + 255) / 256, 256, 0, stream>>>(
      atoms, F[17], F[15], F[14], F[16], F[13], hs, hv);
  k_edge_embed<<<(NEDGES + 255) / 256, 256, 0, stream>>>(
      edge_s, edge_v, F[0], F[1], F[2], F[3], F[4], F[5], F[7], F[6], es, ev);

  // 5 GVPConv layers
  long accN = (long)NNODES * NS + (long)NNODES * 48 + NNODES;
  for (int l = 0; l < 5; ++l) {
    int b = 18 + 34 * l;
    k_zero<<<(int)((accN + 255) / 256), 256, 0, stream>>>(sAcc, accN);

    MsgP mp;
    mp.g0.wh = F[b + 16]; mp.g0.ws_b = m0b[l]; mp.g0.wsv_b = m0gb[l];
    mp.g0.wv = F[b + 21]; mp.g0.ws = m0ws[l]; mp.g0.wsv = m0g[l];
    mp.g1.wh = F[b + 22]; mp.g1.ws_b = m1b[l]; mp.g1.wsv_b = m1gb[l];
    mp.g1.wv = F[b + 27]; mp.g1.ws = m1ws[l]; mp.g1.wsv = m1g[l];
    mp.g2.wh = F[b + 28]; mp.g2.ws_b = m2b[l]; mp.g2.wsv_b = m2gb[l];
    mp.g2.wv = F[b + 33]; mp.g2.ws = m2ws[l]; mp.g2.wsv = m2g[l];
    k_msg<<<NEDGES / 16, 128, 0, stream>>>(mp, hs, hv, es, ev, esrc, edst,
                                           sAcc, vAcc, cnt);

    FFP fp;
    fp.f0.wh = F[b + 0]; fp.f0.ws_b = f0b[l]; fp.f0.wsv_b = f0gb[l];
    fp.f0.wv = F[b + 5]; fp.f0.ws = f0ws[l]; fp.f0.wsv = f0g[l];
    fp.f1.wh = F[b + 6]; fp.f1.ws_b = f1b[l]; fp.f1.wsv_b = f1gb[l];
    fp.f1.wv = F[b + 11]; fp.f1.ws = f1ws[l]; fp.f1.wsv = f1g[l];
    fp.ln0b = F[b + 12]; fp.ln0g = F[b + 13]; fp.ln1b = F[b + 14]; fp.ln1g = F[b + 15];
    k_node<<<NNODES / 16, 128, 0, stream>>>(fp, hs, hv, sAcc, vAcc, cnt);
  }

  // output head + scatter-mean pooling
  long poolN = (long)NGRAPH * ODIM + NGRAPH;
  k_zero<<<(int)((poolN + 255) / 256), 256, 0, stream>>>(gsum, poolN);
  k_out<<<NNODES / 16, 128, 0, stream>>>(F[8], woutws, woutb, F[12], F[11],
                                         hs, hv, batch, gsum, gcnt);
  k_finalize<<<(NGRAPH * ODIM + 255) / 256, 256, 0, stream>>>(gsum, gcnt, out);
}
